// Task_specific_attention_58884001628715
// MI455X (gfx1250) — matrix operand order repaired and checked by reading, not compiled, not run on hardware
//
#include <hip/hip_runtime.h>
#include <hip/hip_bf16.h>

#define Bn 64
#define Sn 2048
#define Hn 512

typedef __attribute__((ext_vector_type(16))) __bf16 v16bf;
typedef __attribute__((ext_vector_type(8)))  float  v8f;

__device__ __forceinline__ unsigned short f2bf(float f) {
    unsigned u = __float_as_uint(f);
    u += 0x7FFFu + ((u >> 16) & 1u);   // round-to-nearest-even
    return (unsigned short)(u >> 16);
}

// ---------------------------------------------------------------------------
// Kernel 0: pack W (f32 row-major [K=H][N=H]) into bf16 WMMA B-fragment order.
// packed[((ntile*16 + ktile)*32 + lane)*16 + e] = bf16(W[ktile*32 + 16*(lane/16) + e][ntile*16 + lane%16])
// ---------------------------------------------------------------------------
__global__ __launch_bounds__(256) void pack_w_kernel(const float* __restrict__ W,
                                                     unsigned short* __restrict__ wb) {
    int idx   = blockIdx.x * 256 + threadIdx.x;   // < H*H = 262144
    int e     = idx & 15;
    int lane  = (idx >> 4) & 31;
    int tile  = idx >> 9;                         // ntile*16 + ktile
    int ktile = tile & 15;
    int ntile = tile >> 4;
    int K = ktile * 32 + ((lane >> 4) << 4) + e;
    int N = ntile * 16 + (lane & 15);
    wb[idx] = f2bf(W[K * Hn + N]);
}

// ---------------------------------------------------------------------------
// Kernel 1: fused  scores[r] = w_attn . tanh(X[r,:] @ W + b)
// 256 threads = 8 waves; 128 rows per WG; bf16 WMMA 16x16x32, f32 accumulate.
// ---------------------------------------------------------------------------
#define APITCH 1040   // (512 + 8) bf16 per row -> bank-conflict-free, 16B aligned
#define A_BYTES (128 * APITCH)     // 133120
#define B_BYTES 16384              // one N-tile W panel: 16 ktiles * 1KB frags
#define SMEM_BYTES (A_BYTES + B_BYTES + 2048 + 2048)

__global__ __launch_bounds__(256) void scores_kernel(
        const float* __restrict__ x, const int* __restrict__ seqlen,
        const unsigned short* __restrict__ wb, const float* __restrict__ wattn,
        const float* __restrict__ bproj, float* __restrict__ scores) {
    extern __shared__ char smem[];
    char*  aLds  = smem;
    char*  bLds  = smem + A_BYTES;
    float* waLds = (float*)(smem + A_BYTES + B_BYTES);
    float* bpLds = (float*)(smem + A_BYTES + B_BYTES + 2048);

    const int t       = threadIdx.x;
    const int rowbase = blockIdx.x * 128;
    const int b       = rowbase >> 11;     // / Sn
    const int s0      = rowbase & (Sn - 1);
    const int len     = seqlen[b];

    // Fully-masked block: softmax weight will be exactly 0 -> skip the GEMM.
    if (s0 >= len) {
        if (t < 128) scores[rowbase + t] = -1e30f;
        return;
    }

    for (int j = t; j < Hn; j += 256) { waLds[j] = wattn[j]; bpLds[j] = bproj[j]; }

    // Load 128 rows of X (f32), convert to bf16 into LDS (row-major, padded pitch).
    const float4* xsrc = (const float4*)(x + (size_t)rowbase * Hn);
    for (int j = t; j < 128 * (Hn / 4); j += 256) {
        float4 f = xsrc[j];
        int row = j >> 7;          // / 128
        int c4  = j & 127;
        unsigned lo = ((unsigned)f2bf(f.y) << 16) | f2bf(f.x);
        unsigned hi = ((unsigned)f2bf(f.w) << 16) | f2bf(f.z);
        *(uint2*)(aLds + row * APITCH + c4 * 8) = make_uint2(lo, hi);
    }

    const int lane = t & 31;
    const int wave = t >> 5;
    const int half = lane >> 4;
    const char* aRow = aLds + (wave * 16 + (lane & 15)) * APITCH;

    float sp[8];
#pragma unroll
    for (int v = 0; v < 8; ++v) sp[v] = 0.f;

    union Frag { uint4 u[2]; v16bf v; };

    for (int ntile = 0; ntile < Hn / 16; ++ntile) {
        __syncthreads();  // also covers the phase-A stores on first iteration
        const uint4* src = (const uint4*)(wb + ntile * 16 * 512);
        for (int j = t; j < B_BYTES / 16; j += 256) ((uint4*)bLds)[j] = src[j];
        __syncthreads();

        v8f acc = {0.f, 0.f, 0.f, 0.f, 0.f, 0.f, 0.f, 0.f};
#pragma unroll
        for (int i = 0; i < 16; ++i) {          // K = 512 in steps of 32
            Frag a, bf;
            a.u[0]  = *(const uint4*)(aRow + 64 * i + 32 * half);
            a.u[1]  = *(const uint4*)(aRow + 64 * i + 32 * half + 16);
            const char* bp = bLds + i * 1024 + lane * 32;
            bf.u[0] = *(const uint4*)(bp);
            bf.u[1] = *(const uint4*)(bp + 16);
            acc = __builtin_amdgcn_wmma_f32_16x16x32_bf16(
                    false, a.v, false, bf.v, (short)0, acc, false, false);
        }
        const int   ncol = ntile * 16 + (lane & 15);
        const float wcol = waLds[ncol];
        const float bcol = bpLds[ncol];
#pragma unroll
        for (int v = 0; v < 8; ++v) sp[v] += tanhf(acc[v] + bcol) * wcol;
    }

    // C layout: VGPR v, lanes 0-15 -> M=v ; lanes 16-31 -> M=v+8. Reduce over N.
#pragma unroll
    for (int v = 0; v < 8; ++v) {
        float s = sp[v];
        s += __shfl_xor(s, 1);
        s += __shfl_xor(s, 2);
        s += __shfl_xor(s, 4);
        s += __shfl_xor(s, 8);
        sp[v] = s;
    }
    if (lane == 0) {
#pragma unroll
        for (int v = 0; v < 8; ++v) scores[rowbase + wave * 16 + v] = sp[v];
    }
    if (lane == 16) {
#pragma unroll
        for (int v = 0; v < 8; ++v) scores[rowbase + wave * 16 + 8 + v] = sp[v];
    }
}

// ---------------------------------------------------------------------------
// Kernel 2: softmax over S, mask, renormalize (exact reference semantics:
// attn = mask * e / (zm + 1e-6 * z),  guarded for len == 0).
// ---------------------------------------------------------------------------
__global__ __launch_bounds__(256) void softmax_kernel(const float* __restrict__ scores,
                                                      const int* __restrict__ seqlen,
                                                      float* __restrict__ attn) {
    __shared__ float red[256];
    const int b = blockIdx.x, t = threadIdx.x;
    const float* srow = scores + (size_t)b * Sn;
    float v[8];
    float m = -3.4e38f;
#pragma unroll
    for (int i = 0; i < 8; ++i) { v[i] = srow[t + 256 * i]; m = fmaxf(m, v[i]); }
    red[t] = m; __syncthreads();
    for (int off = 128; off > 0; off >>= 1) {
        if (t < off) red[t] = fmaxf(red[t], red[t + off]);
        __syncthreads();
    }
    m = red[0]; __syncthreads();

    const int len = seqlen[b];
    float e[8], z = 0.f, zm = 0.f;
#pragma unroll
    for (int i = 0; i < 8; ++i) {
        e[i] = expf(v[i] - m);
        z += e[i];
        if (t + 256 * i < len) zm += e[i];
    }
    red[t] = z; __syncthreads();
    for (int off = 128; off > 0; off >>= 1) { if (t < off) red[t] += red[t + off]; __syncthreads(); }
    z = red[0]; __syncthreads();
    red[t] = zm; __syncthreads();
    for (int off = 128; off > 0; off >>= 1) { if (t < off) red[t] += red[t + off]; __syncthreads(); }
    zm = red[0];

    const float inv = (zm > 0.f) ? 1.f / (zm + 1e-6f * z) : 0.f;
    float* arow = attn + (size_t)b * Sn;
#pragma unroll
    for (int i = 0; i < 8; ++i) {
        int s = t + 256 * i;
        arow[s] = (s < len) ? e[i] * inv : 0.f;
    }
}

// ---------------------------------------------------------------------------
// Kernels 3+4: zero output, then weighted pooling with atomic f32 adds.
// Rows with attn==0 (masked) skip the X read entirely.
// ---------------------------------------------------------------------------
__global__ __launch_bounds__(256) void zero_out_kernel(float* __restrict__ out, int n) {
    int i = blockIdx.x * 256 + threadIdx.x;
    if (i < n) out[i] = 0.f;
}

__global__ __launch_bounds__(256) void pool_kernel(const float* __restrict__ x,
                                                   const float* __restrict__ attn,
                                                   float* __restrict__ out) {
    const int b = blockIdx.x;
    const int chunk = blockIdx.y;         // 16 chunks of S/16 = 128
    const int t = threadIdx.x;
    const int SC = Sn / 16;
    const float* xb = x + ((size_t)b * Sn + (size_t)chunk * SC) * Hn;
    const float* ab = attn + (size_t)b * Sn + (size_t)chunk * SC;
    float a0 = 0.f, a1 = 0.f;
    for (int s = 0; s < SC; ++s) {
        float w = ab[s];
        if (w != 0.f) {                    // uniform branch across the block
            a0 = fmaf(xb[(size_t)s * Hn + t], w, a0);
            a1 = fmaf(xb[(size_t)s * Hn + t + 256], w, a1);
        }
    }
    atomicAdd(&out[b * Hn + t], a0);
    atomicAdd(&out[b * Hn + t + 256], a1);
}

// ---------------------------------------------------------------------------
extern "C" void kernel_launch(void* const* d_in, const int* in_sizes, int n_in,
                              void* d_out, int out_size, void* d_ws, size_t ws_size,
                              hipStream_t stream) {
    (void)in_sizes; (void)n_in; (void)ws_size;
    const float* x      = (const float*)d_in[0];   // [B,S,H] f32
    const int*   seqlen = (const int*)  d_in[1];   // [B]
    const float* W      = (const float*)d_in[2];   // [H,H]
    const float* bproj  = (const float*)d_in[3];   // [H]
    const float* wattn  = (const float*)d_in[4];   // [H,1]

    unsigned short* wb  = (unsigned short*)d_ws;                    // 512 KB bf16 packed W
    float* scores       = (float*)((char*)d_ws + 512 * 1024);       // 512 KB
    float* attn         = (float*)((char*)d_ws + 1024 * 1024);      // 512 KB
    float* out          = (float*)d_out;

    pack_w_kernel<<<(Hn * Hn) / 256, 256, 0, stream>>>(W, wb);
    scores_kernel<<<(Bn * Sn) / 128, 256, SMEM_BYTES, stream>>>(x, seqlen, wb, wattn, bproj, scores);
    softmax_kernel<<<Bn, 256, 0, stream>>>(scores, seqlen, attn);
    zero_out_kernel<<<(out_size + 255) / 256, 256, 0, stream>>>(out, out_size);
    pool_kernel<<<dim3(Bn, 16), 256, 0, stream>>>(x, attn, out);
}
